// PETNNCell_24919400252151
// MI455X (gfx1250) — matrix operand order repaired and checked by reading, not compiled, not run on hardware
//
#include <hip/hip_runtime.h>

// ---------------------------------------------------------------------------
// PETNN cell, fully fused for MI455X (gfx1250, wave32, WMMA bf16 + TDM).
//
//   B=16384, IN=HID=CELL=512, CONCAT=1024
//   GEMM work ~60 GFLOP; HBM floor ~170MB -> ~7.3us @23.3TB/s.
//   bf16 WMMA (16x16x32, f32 acc) keeps compute ~balanced with memory.
//
//   Factorization: h = sigmoid(Gx + C_scale*Gs + b_h) with
//     Gx = X @ Wh[:, :512].T,  Gs = S_prev @ Wh[:, 512:].T
//   -> all 7 GEMM passes are independent of the scalar (T_t/m/C_scale) path,
//      so one kernel computes everything.
//
//   B-panels are pre-packed fragment-linear in d_ws so each pass's 32KB panel
//   is ONE contiguous block -> staged by a single Tensor Data Mover descriptor
//   (tensor_load_to_lds, TENSORcnt) instead of per-thread load+ds_store.
// ---------------------------------------------------------------------------

#define BROWS 16384
#define KD    512
#define MT    128          // rows per workgroup (8 waves x 16)

typedef __attribute__((ext_vector_type(16))) __bf16        v16bf;
typedef __attribute__((ext_vector_type(8)))  float         v8f;
typedef __attribute__((ext_vector_type(8)))  unsigned int  v8u;
typedef __attribute__((ext_vector_type(4)))  unsigned int  v4u;
typedef __attribute__((ext_vector_type(8)))  int           v8i;
typedef __attribute__((ext_vector_type(4)))  int           v4i;

#if __has_builtin(__builtin_amdgcn_tensor_load_to_lds)
#define HAVE_TDM 1
#else
#define HAVE_TDM 0
#endif

__device__ __forceinline__ unsigned short f2bf(float f) {
  unsigned u = __builtin_bit_cast(unsigned, f);
  u += 0x7FFFu + ((u >> 16) & 1u);              // round-to-nearest-even
  return (unsigned short)(u >> 16);
}
__device__ __forceinline__ float bf2f(unsigned short h) {
  unsigned u = ((unsigned)h) << 16;
  return __builtin_bit_cast(float, u);
}
__device__ __forceinline__ float sigmoidf(float x) {
  return 1.0f / (1.0f + __expf(-x));
}
// Load one 16x16x32 bf16 fragment: lane's 16 bf16 live at p + lane*16 shorts.
__device__ __forceinline__ v16bf ld_frag(const unsigned short* p) {
  const uint4* q = (const uint4*)p;
  uint4 a = q[0]; uint4 b = q[1];
  v8u r; r[0]=a.x; r[1]=a.y; r[2]=a.z; r[3]=a.w; r[4]=b.x; r[5]=b.y; r[6]=b.z; r[7]=b.w;
  return __builtin_bit_cast(v16bf, r);
}
__device__ __forceinline__ float wave_sum(float v) {
  v += __shfl_xor(v, 16, 32);
  v += __shfl_xor(v, 8, 32);
  v += __shfl_xor(v, 4, 32);
  v += __shfl_xor(v, 2, 32);
  v += __shfl_xor(v, 1, 32);
  return v;
}

// Issue one 1-D TDM copy: 32KB contiguous global -> LDS (data_size=8B,
// tile_dim0=4096 units, single row). D# per CDNA5 ISA ch.8.
__device__ __forceinline__ void tdm_copy_32k(const void* gsrc, void* lds_dst) {
#if HAVE_TDM
  unsigned lds_off = (unsigned)(size_t)lds_dst;            // low 32b = LDS offset
  unsigned long long ga = (unsigned long long)(size_t)gsrc;
  v4u g0;
  g0[0] = 1u;                                              // count=1, user mode
  g0[1] = lds_off;                                         // lds_addr
  g0[2] = (unsigned)ga;                                    // global_addr[31:0]
  g0[3] = (unsigned)((ga >> 32) & 0x01FFFFFFu) | (2u << 30); // ga[56:32] | type=2
  v8i g1;
  g1[0] = (int)(3u << 16);            // wg_mask=0, data_size=3 (8B), no pad/iter
  g1[1] = (int)(0x1000u << 16);       // tensor_dim0[15:0]=4096 in [31:16]
  g1[2] = (int)(1u << 16);            // tensor_dim0[31:16]=0, tensor_dim1[15:0]=1
  g1[3] = (int)(0x1000u << 16);       // tensor_dim1[31:16]=0, tile_dim0=4096
  g1[4] = 1;                          // tile_dim1=1, tile_dim2=0
  g1[5] = 0x1000;                     // tensor_dim0_stride[31:0]=4096
  g1[6] = (int)(0x1000u << 16);       // stride0[47:32]=0, stride1[15:0]=4096
  g1[7] = 0;                          // stride1[47:16]=0
  v4i gz = {0, 0, 0, 0};              // groups 2/3 unused (<=2D tensor)
#if defined(__clang_major__) && (__clang_major__ >= 23)
  v8i g4 = {0, 0, 0, 0, 0, 0, 0, 0};
  __builtin_amdgcn_tensor_load_to_lds(g0, g1, gz, gz, g4, 0);
#else
  __builtin_amdgcn_tensor_load_to_lds(g0, g1, gz, gz, 0);
#endif
#endif
}

// ---------------------------------------------------------------------------
// Pre-kernel: pack weights fp32 -> bf16 into B-fragment-linear layout in ws.
// Layout: wsB[pass][ntile 0..31][kfrag 0..15][lane 0..31][16 bf16]
//   B-frag (32x16 bf16): lane l -> column n0+(l&15), K = k0 + (l>>4)*8 + {0..7, 16..23}
// Passes: 0 Zc|X  1 Zc|S  2 Zw|X  3 Zw|S  4 It|X  5 Gx(Wh)|X  6 Gs(Wh)|S
// ---------------------------------------------------------------------------
__global__ __launch_bounds__(256) void petnn_pack_weights(
    const float* __restrict__ Wzc, const float* __restrict__ Wzw,
    const float* __restrict__ Wit, const float* __restrict__ Wh,
    unsigned short* __restrict__ wsB) {
  int u = blockIdx.x * 256 + threadIdx.x;        // (p, nt, kf, lane) unit
  if (u >= 7 * 32 * 16 * 32) return;
  int lane = u & 31;
  int kf   = (u >> 5) & 15;
  int nt   = (u >> 9) & 31;
  int p    = u >> 14;
  int n    = nt * 16 + (lane & 15);
  int kb   = kf * 32 + (lane >> 4) * 8;
  const float* src; int stride, koff;
  switch (p) {
    case 0:  src = Wzc; stride = 1024; koff = 0;   break;
    case 1:  src = Wzc; stride = 1024; koff = 512; break;
    case 2:  src = Wzw; stride = 1024; koff = 0;   break;
    case 3:  src = Wzw; stride = 1024; koff = 512; break;
    case 4:  src = Wit; stride = 512;  koff = 0;   break;
    case 5:  src = Wh;  stride = 1024; koff = 0;   break;
    default: src = Wh;  stride = 1024; koff = 512; break;
  }
  const float4* r4 = (const float4*)(src + (size_t)n * stride + koff + kb);
  unsigned short* dst = wsB + (size_t)u * 16;
#pragma unroll
  for (int j = 0; j < 4; ++j) {
    float4 v = r4[j + (j >> 1) * 2];
    dst[j*4+0] = f2bf(v.x); dst[j*4+1] = f2bf(v.y);
    dst[j*4+2] = f2bf(v.z); dst[j*4+3] = f2bf(v.w);
  }
}

// ---------------------------------------------------------------------------
// Fused cell kernel. Grid: 128 WGs x 256 threads (8 waves). Each wave owns a
// 16-row block; all waves share the current 32-column chunk so every B frag
// is staged to LDS exactly once per WG (weights hit L2 once per WG: ~3.5MB).
// ---------------------------------------------------------------------------
__global__ __launch_bounds__(256) void petnn_fused(
    const float* __restrict__ X,   const float* __restrict__ S,
    const float* __restrict__ Cp,  const float* __restrict__ Tp,
    const float* __restrict__ Wzt, const float* __restrict__ bZt,
    const float* __restrict__ bZc, const float* __restrict__ bZw,
    const float* __restrict__ bIt, const float* __restrict__ Wrt,
    const float* __restrict__ bRt, const float* __restrict__ bh,
    const unsigned short* __restrict__ wsB, float* __restrict__ out) {
  // LDS budget: 128KB + 128KB A-frag tiles, 32KB B stage, ~15KB scalars
  __shared__ unsigned short sXf[8 * 16 * 32 * 16];   // 128 KB, A-frag layout
  __shared__ unsigned short sSf[8 * 16 * 32 * 16];   // 128 KB
  __shared__ unsigned short sB [2 * 16 * 32 * 16];   // 32 KB (2 n-tiles)
  __shared__ float sWzt[1024];
  __shared__ float sWrt[512];
  __shared__ float sBZc[512], sBZw[512], sBIt[512], sBH[512];
  __shared__ float sM[MT], sCs[MT];

  const int tid    = threadIdx.x;
  const int lane   = tid & 31;
  const int rb     = tid >> 5;               // wave id == row block (0..7)
  const int wgRow0 = blockIdx.x * MT;

  // ---- stage biases + GEMV weight rows ----
  for (int i = tid; i < 512; i += 256) {
    sBZc[i] = bZc[i]; sBZw[i] = bZw[i]; sBIt[i] = bIt[i];
    sBH[i]  = bh[i];  sWrt[i] = Wrt[i];
  }
  for (int i = tid; i < 1024; i += 256) sWzt[i] = Wzt[i];

  // ---- stage A tiles: fp32 global -> bf16 LDS in WMMA A-frag layout ----
  // A-frag (16x32 bf16): lane = m + 16*((kk>>3)&1), idx = (kk&7)+8*(kk>>4)
  {
    const float*    srcs[2] = { X + (size_t)wgRow0 * KD, S + (size_t)wgRow0 * KD };
    unsigned short* dsts[2] = { sXf, sSf };
#pragma unroll
    for (int ssel = 0; ssel < 2; ++ssel) {
      const float4* g = (const float4*)srcs[ssel];
      unsigned short* d = dsts[ssel];
      for (int it = 0; it < 64; ++it) {
        int f4  = tid + it * 256;            // coalesced float4 index
        int e0  = f4 << 2;
        int row = e0 >> 9;
        int k0  = e0 & 511;
        float4 v = g[f4];
        int rb2 = row >> 4, ml = row & 15, kf = k0 >> 5, kk = k0 & 31;
        int ln   = ml + (((kk >> 3) & 1) << 4);
        int idx0 = (kk & 7) + ((kk >> 4) << 3);   // 4 consecutive idx (kk%8 in {0,4})
        unsigned lo = (unsigned)f2bf(v.x) | ((unsigned)f2bf(v.y) << 16);
        unsigned hi = (unsigned)f2bf(v.z) | ((unsigned)f2bf(v.w) << 16);
        *(uint2*)(d + (((rb2 * 16 + kf) * 32 + ln) << 4) + idx0) = make_uint2(lo, hi);
      }
    }
  }
  __syncthreads();

  // ---- per-row scalar path: Z_t, R_t -> T_t, m, C_scale ----
  {
#pragma unroll 1
    for (int r = 0; r < 16; ++r) {
      int rowL = rb * 16 + r;
      int grow = wgRow0 + rowL;
      float sz = 0.f, sr = 0.f, sc = 0.f;
#pragma unroll 1
      for (int j = 0; j < 16; ++j) {
        int k  = lane + j * 32;
        int kf = k >> 5, kk = k & 31;
        int off = (((rb * 16 + kf) * 32 + (r + (((kk >> 3) & 1) << 4))) << 4)
                  + (kk & 7) + ((kk >> 4) << 3);
        float xv = bf2f(sXf[off]);
        float sv = bf2f(sSf[off]);
        sz += xv * sWzt[k] + sv * sWzt[512 + k];
        sr += xv * sWrt[k];
        sc += Cp[(size_t)grow * KD + k];     // coalesced 128B row read
      }
      sz = wave_sum(sz); sr = wave_sum(sr); sc = wave_sum(sc);
      if (lane == 0) {
        float zt = sz + bZt[0];
        float Rt = sigmoidf(sr + bRt[0]);
        float Tt = Rt * sigmoidf(Tp[grow] + zt) - 1.0f;
        float mF = (Tt <= 0.0f) ? 1.0f : 0.0f;
        out[(size_t)2 * BROWS * KD + grow] = fmaxf(Tt, 0.0f);
        sM[rowL]  = mF;
        sCs[rowL] = (1.0f - mF) * sc * (1.0f / 512.0f);  // mean((1-m)*C_prev)
      }
    }
  }

  const int hl   = lane >> 4;     // C-matrix: row = i + 8*hl, col = lane&15
  const int ncol = lane & 15;

  // ---- 16 column chunks of 32; per chunk: 7 GEMM passes + fused epilogue ----
  for (int nt0 = 0; nt0 < 32; nt0 += 2) {
    v8f acc[5][2];                // Zc, Zw, It, Gx, Gs  x  2 n-tiles
    const v8f vzero = {0.f, 0.f, 0.f, 0.f, 0.f, 0.f, 0.f, 0.f};
#pragma unroll
    for (int a = 0; a < 5; ++a) { acc[a][0] = vzero; acc[a][1] = vzero; }

#pragma unroll
    for (int p = 0; p < 7; ++p) {
      __syncthreads();            // previous consumers of sB are done
#if HAVE_TDM
      // One TDM descriptor per pass: 32KB panel, issued by wave 0 only.
      if (tid < 32) {
        tdm_copy_32k(wsB + ((size_t)(p * 32 + nt0) * 8192), sB);
        __builtin_amdgcn_s_wait_tensorcnt(0);
      }
#else
      {
        const uint4* src = (const uint4*)(wsB + ((size_t)(p * 32 + nt0) * 8192));
        uint4* dst = (uint4*)sB;
#pragma unroll
        for (int j = 0; j < 8; ++j) dst[tid + j * 256] = src[tid + j * 256];
      }
#endif
      if (p < 6)  // gfx1250 global_prefetch_b8: pull next panel toward L2/L0
        __builtin_prefetch(wsB + ((size_t)((p + 1) * 32 + nt0) * 8192), 0, 1);
      __syncthreads();

      const unsigned short* aBase =
          ((p == 1) || (p == 3) || (p == 6)) ? (sSf + rb * 8192) : (sXf + rb * 8192);
      const int ai = (p < 2) ? 0 : (p < 4) ? 1 : (p == 4) ? 2 : (p == 5) ? 3 : 4;

#pragma unroll 4
      for (int kf = 0; kf < 16; ++kf) {
        v16bf a  = ld_frag(aBase + kf * 512 + lane * 16);
        v16bf b0 = ld_frag(sB + kf * 512 + lane * 16);
        acc[ai][0] = __builtin_amdgcn_wmma_f32_16x16x32_bf16(
            false, a, false, b0, (short)0, acc[ai][0], false, false);
        v16bf b1 = ld_frag(sB + 8192 + kf * 512 + lane * 16);
        acc[ai][1] = __builtin_amdgcn_wmma_f32_16x16x32_bf16(
            false, a, false, b1, (short)0, acc[ai][1], false, false);
      }
    }

    // ---- fused epilogue for columns [nt0*16, nt0*16+32) ----
#pragma unroll
    for (int t = 0; t < 2; ++t) {
      int c = nt0 * 16 + t * 16 + ncol;
      float vbZc = sBZc[c], vbZw = sBZw[c], vbIt = sBIt[c], vbH = sBH[c];
#pragma unroll
      for (int i = 0; i < 8; ++i) {
        int rowL = rb * 16 + i + 8 * hl;
        size_t g = (size_t)(wgRow0 + rowL) * KD + c;
        float mF = sM[rowL];
        float cs = sCs[rowL];
        float zc = acc[0][t][i] + vbZc;
        float zw = sigmoidf(acc[1][t][i] + vbZw);
        float it = acc[2][t][i] + vbIt;
        float h  = sigmoidf(acc[3][t][i] + cs * acc[4][t][i] + vbH);
        float cpv = Cp[g], xv = X[g], sp = S[g];
        float cn = (1.0f - mF) * cpv + mF * it + zc;
        float sn = sigmoidf((1.0f - zw) * sp + zw * h + xv);
        out[g] = sn;                               // S_new
        out[(size_t)BROWS * KD + g] = cn;          // C_new
      }
    }
  }
}

// ---------------------------------------------------------------------------
extern "C" void kernel_launch(void* const* d_in, const int* in_sizes, int n_in,
                              void* d_out, int out_size, void* d_ws, size_t ws_size,
                              hipStream_t stream) {
  const float* X   = (const float*)d_in[0];
  const float* S   = (const float*)d_in[1];
  const float* Cp  = (const float*)d_in[2];
  const float* Tp  = (const float*)d_in[3];
  const float* Wzt = (const float*)d_in[4];
  const float* bZt = (const float*)d_in[5];
  const float* Wzc = (const float*)d_in[6];
  const float* bZc = (const float*)d_in[7];
  const float* Wzw = (const float*)d_in[8];
  const float* bZw = (const float*)d_in[9];
  const float* Wit = (const float*)d_in[10];
  const float* bIt = (const float*)d_in[11];
  const float* Wrt = (const float*)d_in[12];
  const float* bRt = (const float*)d_in[13];
  const float* Wh  = (const float*)d_in[14];
  const float* bh  = (const float*)d_in[15];
  unsigned short* wsB = (unsigned short*)d_ws;   // 3.5 MB bf16 packed weights
  float* out = (float*)d_out;

  petnn_pack_weights<<<448, 256, 0, stream>>>(Wzc, Wzw, Wit, Wh, wsB);
  petnn_fused<<<BROWS / MT, 256, 0, stream>>>(X, S, Cp, Tp, Wzt, bZt, bZc, bZw,
                                              bIt, Wrt, bRt, bh, wsB, out);
}
